// AttentionPooling_44238163149017
// MI455X (gfx1250) — compile-verified, matrix operand order checked by reading, not run on hardware
//
#include <hip/hip_runtime.h>
#include <hip/hip_bf16.h>
#include <math.h>

typedef __attribute__((ext_vector_type(16))) __bf16 v16bf;
typedef __attribute__((ext_vector_type(8)))  __bf16 v8bf;
typedef __attribute__((ext_vector_type(8)))  float  v8f;

#define D_DIM 256
#define H_DIM 256
#define NGRAPH 1024
#define SCORE_BLOCKS 512
#define LSE_BLOCKS 512

// Branch-free tanh: native V_TANH_F32 on gfx1250 (probe-confirmed last round).
__device__ __forceinline__ float fast_tanh(float x) {
#if __has_builtin(__builtin_amdgcn_tanhf)
    return __builtin_amdgcn_tanhf(x);
#elif __has_builtin(__builtin_amdgcn_tanh_f32)
    return __builtin_amdgcn_tanh_f32(x);
#else
    const float e = __expf(2.0f * x);
    return 1.0f - 2.0f / (e + 1.0f);
#endif
}

// ---------------------------------------------------------------------------
// Kernel 1: scores[i] = tanh(x_i @ W1 + b1) @ W2 + b2, via bf16 WMMA.
// W1 kept transposed in LDS as bf16 (128 KB; WGP has 320 KB).
// Each wave handles a 16-row tile of x. A is held in registers for all K.
// j-loop kept rolled: unrolling makes the compiler preload B tiles and spill
// to scratch (observed in round 2); rolled, each WMMA is fed directly by two
// ds_load_b128 with pipelined s_wait_dscnt.
// ---------------------------------------------------------------------------
__global__ __launch_bounds__(256) void scores_kernel(
    const float* __restrict__ x, const float* __restrict__ W1,
    const float* __restrict__ b1, const float* __restrict__ W2,
    const float* __restrict__ b2, float* __restrict__ scores,
    int N, int numTiles, int totalWaves)
{
    __shared__ __bf16 w1t[D_DIM * H_DIM];   // [h][k], bf16, 128 KB
    __shared__ float  b1s[H_DIM];
    __shared__ float  w2s[H_DIM];

    const int tid = threadIdx.x;

    // Cooperative load: coalesced global read of W1[k][h], scattered LDS write.
    for (int i = tid; i < D_DIM * H_DIM; i += 256) {
        const int k = i >> 8;
        const int h = i & 255;
        w1t[h * D_DIM + k] = (__bf16)W1[i];
    }
    for (int i = tid; i < H_DIM; i += 256) {
        b1s[i] = b1[i];
        w2s[i] = W2[i];
    }
    __syncthreads();

    const int lane  = tid & 31;
    const int wave  = tid >> 5;
    const int gwave = blockIdx.x * 8 + wave;
    const int khalf = lane >> 4;      // which K-half this lane holds
    const int nlane = lane & 15;      // row (for A) / col (for B,C)
    const float b2v = b2[0];

    for (int tile = gwave; tile < numTiles; tile += totalWaves) {
        const int r0  = tile * 16;
        int row = r0 + nlane;
        if (row >= N) row = N - 1;    // clamp (stores are guarded below)

        // ---- Build A (16x256 tile of x) in bf16 registers, WMMA layout:
        // lane<16 holds K in [kb, kb+8) U [kb+16, kb+24); lane>=16 shifted by 8.
        v16bf A[8];
        const float* xrow = x + (long)row * D_DIM;
        #pragma unroll
        for (int s = 0; s < 8; ++s) {
            const int kb = s * 32;
            const float4 f0 = *(const float4*)(xrow + kb + khalf * 8);
            const float4 f1 = *(const float4*)(xrow + kb + khalf * 8 + 4);
            const float4 f2 = *(const float4*)(xrow + kb + 16 + khalf * 8);
            const float4 f3 = *(const float4*)(xrow + kb + 16 + khalf * 8 + 4);
            v16bf a;
            a[0]=(__bf16)f0.x;  a[1]=(__bf16)f0.y;  a[2]=(__bf16)f0.z;  a[3]=(__bf16)f0.w;
            a[4]=(__bf16)f1.x;  a[5]=(__bf16)f1.y;  a[6]=(__bf16)f1.z;  a[7]=(__bf16)f1.w;
            a[8]=(__bf16)f2.x;  a[9]=(__bf16)f2.y;  a[10]=(__bf16)f2.z; a[11]=(__bf16)f2.w;
            a[12]=(__bf16)f3.x; a[13]=(__bf16)f3.y; a[14]=(__bf16)f3.z; a[15]=(__bf16)f3.w;
            A[s] = a;
        }

        float sp[8];
        #pragma unroll
        for (int v = 0; v < 8; ++v) sp[v] = 0.0f;

        // ---- 16 hidden-column tiles; each = 8 WMMA steps over K=256.
        #pragma unroll 1
        for (int j = 0; j < 16; ++j) {
            const int col   = j * 16 + nlane;
            const float b1v = b1s[col];
            const float w2v = w2s[col];
            const __bf16* bcol = &w1t[col * D_DIM];

            v8f c = {};
            #pragma unroll
            for (int s = 0; s < 8; ++s) {
                const int kb = s * 32;
                const v8bf blo = *(const v8bf*)(bcol + kb + khalf * 8);
                const v8bf bhi = *(const v8bf*)(bcol + kb + 16 + khalf * 8);
                const v16bf b = __builtin_shufflevector(blo, bhi,
                    0,1,2,3,4,5,6,7,8,9,10,11,12,13,14,15);
                c = __builtin_amdgcn_wmma_f32_16x16x32_bf16(
                        false, A[s], false, b, (short)0, c, false, false);
            }
            // Epilogue fused: native v_tanh_f32 + dot with W2 column weights.
            // C layout: element (v, lane) = hidden[row r0+v+8*khalf][col].
            #pragma unroll
            for (int v = 0; v < 8; ++v) {
                const float h = fast_tanh(c[v] + b1v);
                sp[v] += h * w2v;
            }
        }

        // Reduce partials across the 16 lanes of each half (cols 0..15).
        #pragma unroll
        for (int v = 0; v < 8; ++v) {
            float t = sp[v];
            t += __shfl_xor(t, 1, 32);
            t += __shfl_xor(t, 2, 32);
            t += __shfl_xor(t, 4, 32);
            t += __shfl_xor(t, 8, 32);
            sp[v] = t + b2v;
        }
        if (nlane == 0) {
            const int base = r0 + khalf * 8;
            #pragma unroll
            for (int v = 0; v < 8; ++v) {
                if (base + v < N) scores[base + v] = sp[v];
            }
        }
    }
}

// ---------------------------------------------------------------------------
// Kernel 2: per-block streaming log-sum-exp partials (m_b, s_b).
// ---------------------------------------------------------------------------
__global__ __launch_bounds__(256) void lse_partial_kernel(
    const float* __restrict__ scores, float* __restrict__ partials, int N)
{
    __shared__ float red[256];
    const int b = blockIdx.x;
    const int t = threadIdx.x;
    const int per = (N + gridDim.x - 1) / gridDim.x;
    const int s0 = b * per;
    int s1 = s0 + per; if (s1 > N) s1 = N;

    float m = -__builtin_inff();
    for (int i = s0 + t; i < s1; i += 256) m = fmaxf(m, scores[i]);
    red[t] = m; __syncthreads();
    for (int off = 128; off > 0; off >>= 1) {
        if (t < off) red[t] = fmaxf(red[t], red[t + off]);
        __syncthreads();
    }
    m = red[0]; __syncthreads();

    float sum = 0.0f;
    for (int i = s0 + t; i < s1; i += 256) sum += __expf(scores[i] - m);
    red[t] = sum; __syncthreads();
    for (int off = 128; off > 0; off >>= 1) {
        if (t < off) red[t] += red[t + off];
        __syncthreads();
    }
    if (t == 0) { partials[2 * b] = m; partials[2 * b + 1] = red[0]; }
}

// ---------------------------------------------------------------------------
// Kernel 3: merge partials -> global (M, S). Single block, deterministic.
// ---------------------------------------------------------------------------
__global__ __launch_bounds__(512) void lse_final_kernel(
    const float* __restrict__ partials, float* __restrict__ MS, int P)
{
    __shared__ float rm[512];
    __shared__ float rs[512];
    const int t = threadIdx.x;
    const float m = (t < P) ? partials[2 * t] : -__builtin_inff();
    rm[t] = m; __syncthreads();
    for (int off = 256; off > 0; off >>= 1) {
        if (t < off) rm[t] = fmaxf(rm[t], rm[t + off]);
        __syncthreads();
    }
    const float M = rm[0]; __syncthreads();
    float s = 0.0f;
    if (t < P) s = partials[2 * t + 1] * __expf(partials[2 * t] - M);
    rs[t] = s; __syncthreads();
    for (int off = 256; off > 0; off >>= 1) {
        if (t < off) rs[t] += rs[t + off];
        __syncthreads();
    }
    if (t == 0) { MS[0] = M; MS[1] = rs[0]; }
}

// ---------------------------------------------------------------------------
// Kernel 4: scores -> softmax weights in place.
// ---------------------------------------------------------------------------
__global__ __launch_bounds__(256) void weights_kernel(
    float* __restrict__ scores, const float* __restrict__ MS, int N)
{
    const int i = blockIdx.x * 256 + threadIdx.x;
    if (i < N) {
        const float invS = 1.0f / MS[1];
        scores[i] = __expf(scores[i] - MS[0]) * invS;
    }
}

// ---------------------------------------------------------------------------
// Kernel 5: segment boundaries via binary search (batch is sorted).
// ---------------------------------------------------------------------------
__global__ __launch_bounds__(256) void starts_kernel(
    const int* __restrict__ batch, int* __restrict__ starts, int N)
{
    const int g = blockIdx.x * 256 + threadIdx.x;
    if (g <= NGRAPH) {
        int lo = 0, hi = N;
        while (lo < hi) {
            const int mid = (lo + hi) >> 1;
            if (batch[mid] < g) lo = mid + 1; else hi = mid;
        }
        starts[g] = lo;
    }
}

// ---------------------------------------------------------------------------
// Kernel 6: weighted segment-sum pooling. One block per graph; thread t owns
// output dim t. Deterministic sequential accumulation, coalesced x reads.
// ---------------------------------------------------------------------------
__global__ __launch_bounds__(256) void pool_kernel(
    const float* __restrict__ x, const float* __restrict__ w,
    const int* __restrict__ starts, float* __restrict__ out)
{
    const int g = blockIdx.x;
    const int t = threadIdx.x;
    const int i0 = starts[g];
    const int i1 = starts[g + 1];
    float acc = 0.0f;
    for (int i = i0; i < i1; ++i) {
        acc += x[(long)i * D_DIM + t] * w[i];
    }
    out[(long)g * D_DIM + t] = acc;
}

// ---------------------------------------------------------------------------
extern "C" void kernel_launch(void* const* d_in, const int* in_sizes, int n_in,
                              void* d_out, int out_size, void* d_ws, size_t ws_size,
                              hipStream_t stream) {
    const float* x     = (const float*)d_in[0];
    const int*   batch = (const int*)  d_in[1];
    const float* W1    = (const float*)d_in[2];
    const float* b1    = (const float*)d_in[3];
    const float* W2    = (const float*)d_in[4];
    const float* b2    = (const float*)d_in[5];
    float* out = (float*)d_out;

    const int N = in_sizes[0] / D_DIM;
    const int numTiles = (N + 15) / 16;

    // Workspace layout
    char* ws = (char*)d_ws;
    float* scores   = (float*)ws;                                   // N floats
    size_t off = ((size_t)N * sizeof(float) + 15) & ~(size_t)15;
    float* partials = (float*)(ws + off);                           // 2*LSE_BLOCKS
    off += 2 * LSE_BLOCKS * sizeof(float);
    float* MS       = (float*)(ws + off);                           // 2 floats
    off += 2 * sizeof(float);
    int*   starts   = (int*)(ws + off);                             // NGRAPH+1

    const int totalWaves = SCORE_BLOCKS * 8;
    scores_kernel<<<SCORE_BLOCKS, 256, 0, stream>>>(
        x, W1, b1, W2, b2, scores, N, numTiles, totalWaves);

    lse_partial_kernel<<<LSE_BLOCKS, 256, 0, stream>>>(scores, partials, N);
    lse_final_kernel<<<1, 512, 0, stream>>>(partials, MS, LSE_BLOCKS);
    weights_kernel<<<(N + 255) / 256, 256, 0, stream>>>(scores, MS, N);
    starts_kernel<<<(NGRAPH + 1 + 255) / 256, 256, 0, stream>>>(batch, starts, N);
    pool_kernel<<<NGRAPH, 256, 0, stream>>>(x, scores, starts, out);
}
